// SharedQkvMultiHeadSelfAttention_31301721653747
// MI455X (gfx1250) — matrix operand order repaired and checked by reading, not compiled, not run on hardware
//
#include <hip/hip_runtime.h>
#include <hip/hip_fp16.h>

typedef __attribute__((ext_vector_type(16))) _Float16     v16h;
typedef __attribute__((ext_vector_type(8)))  float        v8f;
typedef __attribute__((ext_vector_type(4)))  unsigned int u32x4;
typedef __attribute__((ext_vector_type(4)))  float        f32x4;

union AFrag { v16h v; unsigned int u[8]; };
union BFrag { v16h v; u32x4 q[2]; };
union H8    { u32x4 q; _Float16 h[8]; };
union HP    { unsigned int u; _Float16 h[2]; };

__device__ __forceinline__ v8f wmma_f16(v16h a, v16h b, v8f c) {
  // D = A(16x32 f16) * B(32x16 f16) + C(16x16 f32)
  return __builtin_amdgcn_wmma_f32_16x16x32_f16(false, a, false, b, (short)0, c,
                                                false, false);
}

// C[m][n] = sum_k A[m][k] * W[n][k] + bias[n]   (W row-major [N][K] = B^T layout)
// A is fp32 or fp16 global; W/bias fp32; output fp16 or fp32.
// Double-buffered: next k-tile is prefetched into registers while WMMAs run.
template <typename AT, typename OT>
__global__ __launch_bounds__(256) void gemm_xwt_bias(
    const AT* __restrict__ A, const float* __restrict__ W,
    const float* __restrict__ bias, OT* __restrict__ C, int M, int N, int K) {
  constexpr int BM = 128, BN = 128, BK = 32, LDT = 40;  // padded f16 stride
  __shared__ _Float16 As[BM * LDT];
  __shared__ _Float16 Bs[BN * LDT];

  const int tid = threadIdx.x;
  const int mBase = blockIdx.y * BM;
  const int nBase = blockIdx.x * BN;
  const int w = tid >> 5, lane = tid & 31;
  const int hf = lane >> 4, l16 = lane & 15;
  const int m0 = (w >> 1) * 32, n0 = (w & 1) * 64;  // 4x2 wave grid, 32x64 per wave

  v8f acc[2][4];
#pragma unroll
  for (int g = 0; g < 2; ++g)
#pragma unroll
    for (int nt = 0; nt < 4; ++nt)
#pragma unroll
      for (int r = 0; r < 8; ++r) acc[g][nt][r] = 0.f;

  // register prefetch buffers
  f32x4 pa[4];   // A tile, fp32 path
  u32x4 pah[2];  // A tile, fp16 path
  f32x4 pb[4];   // W tile (always fp32)

  auto gload = [&](int k0) {
    if constexpr (sizeof(AT) == 4) {
#pragma unroll
      for (int it = 0; it < 4; ++it) {
        int f = tid + it * 256;             // 1024 float4 chunks = 128x32
        int row = f >> 3, col = (f & 7) * 4;
        pa[it] = *(const f32x4*)(&A[(size_t)(mBase + row) * K + k0 + col]);
      }
    } else {
#pragma unroll
      for (int it = 0; it < 2; ++it) {
        int c = tid + it * 256;             // 512 chunks of 8 halves
        int row = c >> 2, col = (c & 3) * 8;
        pah[it] = *(const u32x4*)(&A[(size_t)(mBase + row) * K + k0 + col]);
      }
    }
#pragma unroll
    for (int it = 0; it < 4; ++it) {
      int f = tid + it * 256;
      int row = f >> 3, col = (f & 7) * 4;
      pb[it] = *(const f32x4*)(&W[(size_t)(nBase + row) * K + k0 + col]);
    }
  };

  auto lstore = [&]() {
    if constexpr (sizeof(AT) == 4) {
#pragma unroll
      for (int it = 0; it < 4; ++it) {
        int f = tid + it * 256;
        int row = f >> 3, col = (f & 7) * 4;
        _Float16* d = &As[row * LDT + col];
        d[0] = (_Float16)pa[it].x; d[1] = (_Float16)pa[it].y;
        d[2] = (_Float16)pa[it].z; d[3] = (_Float16)pa[it].w;
      }
    } else {
#pragma unroll
      for (int it = 0; it < 2; ++it) {
        int c = tid + it * 256;
        int row = c >> 2, col = (c & 3) * 8;
        *(u32x4*)(&As[row * LDT + col]) = pah[it];
      }
    }
#pragma unroll
    for (int it = 0; it < 4; ++it) {
      int f = tid + it * 256;
      int row = f >> 3, col = (f & 7) * 4;
      _Float16* d = &Bs[row * LDT + col];
      d[0] = (_Float16)pb[it].x; d[1] = (_Float16)pb[it].y;
      d[2] = (_Float16)pb[it].z; d[3] = (_Float16)pb[it].w;
    }
  };

  gload(0);
  lstore();
  __syncthreads();

  for (int k0 = 0; k0 < K; k0 += BK) {
    const bool hasNext = (k0 + BK) < K;
    if (hasNext) gload(k0 + BK);  // overlap with WMMAs below

    AFrag a[2];
#pragma unroll
    for (int g = 0; g < 2; ++g) {
      const _Float16* base = &As[(m0 + g * 16 + l16) * LDT];
#pragma unroll
      for (int j = 0; j < 8; ++j) {
        int kk = 2 * j + hf * 16;  // contiguous k = 16*half + s (matches B)
        a[g].u[j] = *(const unsigned int*)(base + kk);
      }
    }
#pragma unroll
    for (int nt = 0; nt < 4; ++nt) {
      BFrag b;
      const _Float16* base = &Bs[(n0 + nt * 16 + l16) * LDT + hf * 16];
      b.q[0] = *(const u32x4*)(base);
      b.q[1] = *(const u32x4*)(base + 8);
#pragma unroll
      for (int g = 0; g < 2; ++g) acc[g][nt] = wmma_f16(a[g].v, b.v, acc[g][nt]);
    }
    __syncthreads();           // LDS reads done before overwrite
    if (hasNext) lstore();
    __syncthreads();           // staged tile visible
  }

  // ---- epilogue: bias + store ----
#pragma unroll
  for (int nt = 0; nt < 4; ++nt) {
    int coln = nBase + n0 + nt * 16 + l16;
    float bv = bias[coln];
#pragma unroll
    for (int g = 0; g < 2; ++g)
#pragma unroll
      for (int r = 0; r < 8; ++r) {
        int row = mBase + m0 + g * 16 + hf * 8 + r;  // C/D layout: M = r + half*8
        C[(size_t)row * N + coln] = (OT)(acc[g][nt][r] + bv);
      }
  }
}

// Flash attention over one (batch, head). Vh: f16 [b][n][h*dh] (= projected V).
// Oh: f16 [b][n][h*dh] concat-heads output. Scale folded into Q fragments.
__global__ __launch_bounds__(256) void flash_attn_kernel(
    const _Float16* __restrict__ Vh, _Float16* __restrict__ Oh) {
  constexpr int N = 2048, D = 1024, DH = 64, KT = 64, LDS_K = 72;
  __shared__ _Float16 Kt[KT * LDS_K];       // key tile, row-major [key][dh]
  __shared__ _Float16 Vt[DH * LDS_K];       // value tile, transposed [dh][key]
  __shared__ _Float16 Pw[8 * 16 * LDS_K];   // per-wave P staging [16 rows][64 keys]

  const int tid = threadIdx.x;
  const int w = tid >> 5, lane = tid & 31;
  const int hf = lane >> 4, l16 = lane & 15;
  const int h = blockIdx.y, b = blockIdx.z;
  const int qbase = blockIdx.x * 128 + w * 16;

  const _Float16* Vhead = Vh + (size_t)b * N * D + h * DH;

  // Q fragments, pre-scaled by 1/sqrt(dh) = 0.125 (exact in f16)
  AFrag qf[2];
  {
    const _Float16 qs = (_Float16)0.125f;
    const _Float16* qptr = Vhead + (size_t)(qbase + l16) * D;
#pragma unroll
    for (int ks = 0; ks < 2; ++ks)
#pragma unroll
      for (int j = 0; j < 8; ++j) {
        int kk = 2 * j + hf * 16 + ks * 32;
        HP t;
        t.u = *(const unsigned int*)(qptr + kk);
        t.h[0] *= qs; t.h[1] *= qs;
        qf[ks].u[j] = t.u;
      }
  }

  float m_st[8], l_st[8];
  v8f O[4];
#pragma unroll
  for (int r = 0; r < 8; ++r) { m_st[r] = -1e30f; l_st[r] = 0.f; }
#pragma unroll
  for (int dt = 0; dt < 4; ++dt)
#pragma unroll
    for (int r = 0; r < 8; ++r) O[dt][r] = 0.f;

  // K/V tile register prefetch
  H8 pk[2];
  auto gloadKV = [&](int kb) {
#pragma unroll
    for (int it = 0; it < 2; ++it) {
      int c = tid + it * 256;               // 512 chunks of 8 halves = 64x64
      int row = c >> 3, col = (c & 7) * 8;
      pk[it].q = *(const u32x4*)(Vhead + (size_t)(kb + row) * D + col);
    }
  };
  auto lstoreKV = [&]() {
#pragma unroll
    for (int it = 0; it < 2; ++it) {
      int c = tid + it * 256;
      int row = c >> 3, col = (c & 7) * 8;
      *(u32x4*)(&Kt[row * LDS_K + col]) = pk[it].q;
#pragma unroll
      for (int j = 0; j < 8; ++j) Vt[(col + j) * LDS_K + row] = pk[it].h[j];
    }
  };

  gloadKV(0);
  lstoreKV();
  __syncthreads();

  for (int kb = 0; kb < N; kb += KT) {
    const bool hasNext = (kb + KT) < N;
    if (hasNext) gloadKV(kb + KT);  // overlap with S + softmax below

    // ---- S = (Q*scale) K^T : 4 key sub-tiles x (2 WMMAs over dh=64) ----
    v8f S[4];
#pragma unroll
    for (int nt = 0; nt < 4; ++nt) {
      v8f s;
#pragma unroll
      for (int r = 0; r < 8; ++r) s[r] = 0.f;
#pragma unroll
      for (int ks = 0; ks < 2; ++ks) {
        BFrag kf;
        const _Float16* base = &Kt[(nt * 16 + l16) * LDS_K + ks * 32 + hf * 16];
        kf.q[0] = *(const u32x4*)(base);
        kf.q[1] = *(const u32x4*)(base + 8);
        s = wmma_f16(qf[ks].v, kf.v, s);
      }
      S[nt] = s;
    }

    // ---- online softmax (row reductions across the 16-lane half) + write P ----
    _Float16* prow = &Pw[(w * 16 + hf * 8) * LDS_K + l16];
#pragma unroll
    for (int r = 0; r < 8; ++r) {
      float sv0 = S[0][r], sv1 = S[1][r], sv2 = S[2][r], sv3 = S[3][r];
      float tmx = fmaxf(fmaxf(sv0, sv1), fmaxf(sv2, sv3));
#pragma unroll
      for (int mk = 1; mk < 16; mk <<= 1) tmx = fmaxf(tmx, __shfl_xor(tmx, mk, 32));
      float mnew = fmaxf(m_st[r], tmx);
      float alpha = __expf(m_st[r] - mnew);
      float p0 = __expf(sv0 - mnew), p1 = __expf(sv1 - mnew);
      float p2 = __expf(sv2 - mnew), p3 = __expf(sv3 - mnew);
      float rs = (p0 + p1) + (p2 + p3);
#pragma unroll
      for (int mk = 1; mk < 16; mk <<= 1) rs += __shfl_xor(rs, mk, 32);
      l_st[r] = l_st[r] * alpha + rs;
      m_st[r] = mnew;
#pragma unroll
      for (int dt = 0; dt < 4; ++dt) O[dt][r] *= alpha;
      prow[r * LDS_K +  0] = (_Float16)p0;
      prow[r * LDS_K + 16] = (_Float16)p1;
      prow[r * LDS_K + 32] = (_Float16)p2;
      prow[r * LDS_K + 48] = (_Float16)p3;
    }
    __syncthreads();   // P visible; all Kt reads complete

    // ---- O += P V : re-read P as A-fragments, B from transposed Vt ----
    AFrag pf[2];
    {
      const _Float16* pbase = &Pw[(w * 16 + l16) * LDS_K];
#pragma unroll
      for (int ks = 0; ks < 2; ++ks)
#pragma unroll
        for (int j = 0; j < 8; ++j) {
          int kk = 2 * j + hf * 16 + ks * 32;
          pf[ks].u[j] = *(const unsigned int*)(pbase + kk);
        }
    }
#pragma unroll
    for (int dt = 0; dt < 4; ++dt)
#pragma unroll
      for (int ks = 0; ks < 2; ++ks) {
        BFrag vf;
        const _Float16* base = &Vt[(dt * 16 + l16) * LDS_K + ks * 32 + hf * 16];
        vf.q[0] = *(const u32x4*)(base);
        vf.q[1] = *(const u32x4*)(base + 8);
        O[dt] = wmma_f16(pf[ks].v, vf.v, O[dt]);
      }
    __syncthreads();   // Vt reads complete before overwrite
    if (hasNext) lstoreKV();
    __syncthreads();   // next tile visible
  }

  // ---- normalize and store concat-heads output (f16) ----
  _Float16* outp = Oh + (size_t)b * N * D + h * DH;
#pragma unroll
  for (int dt = 0; dt < 4; ++dt) {
    int dcol = dt * 16 + l16;
#pragma unroll
    for (int r = 0; r < 8; ++r) {
      int row = qbase + hf * 8 + r;
      outp[(size_t)row * D + dcol] = (_Float16)(O[dt][r] / l_st[r]);
    }
  }
}

extern "C" void kernel_launch(void* const* d_in, const int* in_sizes, int n_in,
                              void* d_out, int out_size, void* d_ws, size_t ws_size,
                              hipStream_t stream) {
  (void)in_sizes; (void)n_in; (void)out_size; (void)ws_size;
  const float* x   = (const float*)d_in[0];
  const float* v_w = (const float*)d_in[1];
  const float* v_b = (const float*)d_in[2];
  const float* o_w = (const float*)d_in[3];
  const float* o_b = (const float*)d_in[4];
  float* out = (float*)d_out;

  const int B = 4, N = 2048, D = 1024;
  const int M = B * N;  // 8192

  _Float16* Vh    = (_Float16*)d_ws;                 // 16 MB
  _Float16* AttnH = Vh + (size_t)M * D;              // 16 MB

  dim3 ggrid(D / 128, M / 128);  // (8, 64)
  // 1) V = X * v_w^T + v_b  (fp32 in -> f16 out)
  gemm_xwt_bias<float, _Float16><<<ggrid, 256, 0, stream>>>(x, v_w, v_b, Vh, M, D, D);
  // 2) per-head flash attention: softmax(V V^T / 8) V  (f16 in/out)
  flash_attn_kernel<<<dim3(N / 128, 16, B), 256, 0, stream>>>(Vh, AttnH);
  // 3) out = Attn * o_w^T + o_b  (f16 in -> fp32 out)
  gemm_xwt_bias<_Float16, float><<<ggrid, 256, 0, stream>>>(AttnH, o_w, o_b, out, M, D, D);
}
